// Block_15144054686549
// MI455X (gfx1250) — compile-verified, matrix operand order checked by reading, not run on hardware
//
#include <hip/hip_runtime.h>

typedef _Float16 f16;
typedef __attribute__((ext_vector_type(8)))  _Float16 h8;
typedef __attribute__((ext_vector_type(16))) _Float16 v16h;
typedef __attribute__((ext_vector_type(8)))  float    v8f;

union Frag16 { v16h v; h8 h[2]; };

#define DIMC 768
#define SEQN 1024
#define NHEAD 12
#define HD 64
#define TC 2304
#define HIDDEN 3072
#define MROWS 8192  // B*N

// ---------------------------------------------------------------------------
// LayerNorm: one 256-thread block per row of 768; output f16.
// ---------------------------------------------------------------------------
__global__ __launch_bounds__(256)
void ln_kernel(const float* __restrict__ x, const float* __restrict__ g,
               const float* __restrict__ be, f16* __restrict__ out) {
  const int row = blockIdx.x;
  const float* xr = x + (size_t)row * DIMC;
  const int tid = threadIdx.x;
  float v0 = xr[tid], v1 = xr[tid + 256], v2 = xr[tid + 512];
  float s  = v0 + v1 + v2;
  float ss = v0 * v0 + v1 * v1 + v2 * v2;
#pragma unroll
  for (int m = 1; m < 32; m <<= 1) {
    s  += __shfl_xor(s,  m, 32);
    ss += __shfl_xor(ss, m, 32);
  }
  __shared__ float red[2][8];
  const int wave = tid >> 5, lane = tid & 31;
  if (lane == 0) { red[0][wave] = s; red[1][wave] = ss; }
  __syncthreads();
  if (wave == 0) {
    float a  = (lane < 8) ? red[0][lane] : 0.0f;
    float b2 = (lane < 8) ? red[1][lane] : 0.0f;
#pragma unroll
    for (int m = 1; m < 8; m <<= 1) {
      a  += __shfl_xor(a,  m, 32);
      b2 += __shfl_xor(b2, m, 32);
    }
    if (lane == 0) { red[0][0] = a; red[1][0] = b2; }
  }
  __syncthreads();
  const float mean = red[0][0] * (1.0f / DIMC);
  const float var  = red[1][0] * (1.0f / DIMC) - mean * mean;
  const float rstd = rsqrtf(var + 1e-5f);
  f16* o = out + (size_t)row * DIMC;
  o[tid]       = (f16)((v0 - mean) * rstd * g[tid]       + be[tid]);
  o[tid + 256] = (f16)((v1 - mean) * rstd * g[tid + 256] + be[tid + 256]);
  o[tid + 512] = (f16)((v2 - mean) * rstd * g[tid + 512] + be[tid + 512]);
}

// ---------------------------------------------------------------------------
// WMMA GEMM: C(M,N) = A(M,K)[f16, row-major] @ W(K,N)[f32, row-major] + bias
// Optional exact GELU, optional fp32 residual, f16 or f32 output.
// Block tile 128x128, BK=32; 8 waves (2 in M x 4 in N); wave tile 64x32.
// LDS double-buffered + software pipelined: next tile's global loads are
// issued before the WMMA work on the current tile; one barrier per K-step.
// W is converted f32->f16 and stored N-major in LDS so B-fragments are
// contiguous ds_load_b128's matching the ISA 16-bit A/B VGPR layout.
// ---------------------------------------------------------------------------
template<bool GELU, bool OUTF16, bool HASRES>
__global__ __launch_bounds__(256)
void gemm_wmma(const f16* __restrict__ A, const float* __restrict__ W,
               const float* __restrict__ bias, const float* __restrict__ res,
               f16* __restrict__ outh, float* __restrict__ outf,
               int M, int N, int K) {
  const int AST = 40;   // LDS stride in halves (80B, 16B-aligned, padded)
  const int BST = 40;
  __shared__ f16 As[2][128 * AST];
  __shared__ f16 Bs[2][128 * BST];

  const int tid  = threadIdx.x;
  const int lane = tid & 31;
  const int wave = tid >> 5;
  const int wm = wave >> 2;          // 0..1  (M direction)
  const int wn = wave & 3;           // 0..3  (N direction)
  const int bm = blockIdx.y * 128;
  const int bn = blockIdx.x * 128;
  const int lmod = lane & 15;
  const int lhi  = lane >> 4;
  const int klo  = lhi * 8;

  v8f acc[4][2];
#pragma unroll
  for (int ms = 0; ms < 4; ++ms)
#pragma unroll
    for (int ns = 0; ns < 2; ++ns)
#pragma unroll
      for (int v = 0; v < 8; ++v) acc[ms][ns][v] = 0.0f;

  uint4  areg[2];
  float4 breg[4];

  auto load_regs = [&](int k0) {
#pragma unroll
    for (int i = 0; i < 2; ++i) {                 // A: 512 16B chunks
      const int c = tid + 256 * i;
      const int r = c >> 2, ch = c & 3;
      areg[i] = *(const uint4*)(A + (size_t)(bm + r) * K + k0 + ch * 8);
    }
#pragma unroll
    for (int i = 0; i < 4; ++i) {                 // W: 1024 float4 chunks
      const int c  = tid + 256 * i;
      const int kk = c >> 5, n4 = c & 31;
      breg[i] = *(const float4*)(W + (size_t)(k0 + kk) * N + bn + n4 * 4);
    }
  };
  auto store_lds = [&](int buf) {
#pragma unroll
    for (int i = 0; i < 2; ++i) {
      const int c = tid + 256 * i;
      const int r = c >> 2, ch = c & 3;
      *(uint4*)&As[buf][r * AST + ch * 8] = areg[i];
    }
#pragma unroll
    for (int i = 0; i < 4; ++i) {
      const int c  = tid + 256 * i;
      const int kk = c >> 5, n4 = c & 31;
      Bs[buf][(n4 * 4 + 0) * BST + kk] = (f16)breg[i].x;
      Bs[buf][(n4 * 4 + 1) * BST + kk] = (f16)breg[i].y;
      Bs[buf][(n4 * 4 + 2) * BST + kk] = (f16)breg[i].z;
      Bs[buf][(n4 * 4 + 3) * BST + kk] = (f16)breg[i].w;
    }
  };

  load_regs(0);
  store_lds(0);
  __syncthreads();

  const int nk = K >> 5;                          // K / 32
  for (int kt = 0; kt < nk; ++kt) {
    const int cur = kt & 1, nxt = cur ^ 1;
    if (kt + 1 < nk) {
      load_regs((kt + 1) * 32);                   // overlap with WMMA below
      if (kt + 2 < nk) {                          // pull k+2 toward L2/L0
        __builtin_prefetch(A + (size_t)(bm + (tid >> 2)) * K + (kt + 2) * 32, 0, 0);
        __builtin_prefetch(W + (size_t)((kt + 2) * 32 + (tid >> 5)) * N + bn, 0, 0);
      }
    }

    Frag16 b[2];
#pragma unroll
    for (int ns = 0; ns < 2; ++ns) {
      const int n = wn * 32 + ns * 16 + lmod;
      b[ns].h[0] = *(const h8*)&Bs[cur][n * BST + klo];
      b[ns].h[1] = *(const h8*)&Bs[cur][n * BST + klo + 16];
    }
#pragma unroll
    for (int ms = 0; ms < 4; ++ms) {
      Frag16 a;
      const int r = wm * 64 + ms * 16 + lmod;
      a.h[0] = *(const h8*)&As[cur][r * AST + klo];
      a.h[1] = *(const h8*)&As[cur][r * AST + klo + 16];
#pragma unroll
      for (int ns = 0; ns < 2; ++ns)
        acc[ms][ns] = __builtin_amdgcn_wmma_f32_16x16x32_f16(
            false, a.v, false, b[ns].v, (short)0, acc[ms][ns], false, false);
    }

    if (kt + 1 < nk) store_lds(nxt);
    __syncthreads();
  }

  // Epilogue: D layout is M = v + 8*lhi, N = lane%16 within each 16x16 frag.
#pragma unroll
  for (int ms = 0; ms < 4; ++ms)
#pragma unroll
    for (int ns = 0; ns < 2; ++ns)
#pragma unroll
      for (int v = 0; v < 8; ++v) {
        const int row = bm + wm * 64 + ms * 16 + v + lhi * 8;
        const int col = bn + wn * 32 + ns * 16 + lmod;
        float val = acc[ms][ns][v] + bias[col];
        if (GELU)   val = 0.5f * val * (1.0f + erff(val * 0.70710678f));
        if (HASRES) val += res[(size_t)row * N + col];
        if (OUTF16) outh[(size_t)row * N + col] = (f16)val;
        else        outf[(size_t)row * N + col] = val;
      }
}

// ---------------------------------------------------------------------------
// Flash attention: grid (N/64, B*H), 128 threads = 4 waves.
// Wave w owns 16 query rows; loops over 32-key tiles; QK^T and PV via WMMA.
// qkv layout: [(b*N+n)*2304 + s*768 + h*64 + d], s in {q,k,v}.
// 1/sqrt(hd) is folded into the Q fragment once. The P round-trip through
// LDS is wave-private, so it needs only in-wave DS ordering (s_wait_dscnt),
// not a workgroup barrier.
// ---------------------------------------------------------------------------
__global__ __launch_bounds__(128)
void attn_kernel(const f16* __restrict__ qkv, f16* __restrict__ out) {
  const int KST = 72, VST = 40, PST = 40;   // all 16B-aligned strides
  __shared__ f16 Ks[32 * KST];              // [key][d]  (Bt for QK^T)
  __shared__ f16 Vs[HD * VST];              // [d][key]  (Bt for PV)
  __shared__ f16 Ps[4][16 * PST];           // per-wave P tile [m][key]

  const int bh = blockIdx.y;
  const int b = bh / NHEAD, h = bh % NHEAD;
  const int wave = threadIdx.x >> 5;
  const int lane = threadIdx.x & 31;
  const int lmod = lane & 15, lhi = lane >> 4;
  const int klo  = lhi * 8;
  const int qbase = blockIdx.x * 64 + wave * 16;
  const size_t base = (size_t)b * SEQN * TC + (size_t)h * HD;

  // Q fragments (16 rows x 64 d), pre-scaled by 1/sqrt(64) = 0.125.
  Frag16 q0, q1;
  {
    const f16* qrow = qkv + base + (size_t)(qbase + lmod) * TC;
    q0.h[0] = *(const h8*)(qrow + klo);
    q0.h[1] = *(const h8*)(qrow + klo + 16);
    q1.h[0] = *(const h8*)(qrow + 32 + klo);
    q1.h[1] = *(const h8*)(qrow + 32 + klo + 16);
#pragma unroll
    for (int i = 0; i < 16; ++i) {
      q0.v[i] = q0.v[i] * (_Float16)0.125f;
      q1.v[i] = q1.v[i] * (_Float16)0.125f;
    }
  }

  v8f o[4];
  float mrow[8], lrow[8];
#pragma unroll
  for (int os = 0; os < 4; ++os)
#pragma unroll
    for (int v = 0; v < 8; ++v) o[os][v] = 0.0f;
#pragma unroll
  for (int v = 0; v < 8; ++v) { mrow[v] = -3.0e38f; lrow[v] = 0.0f; }

  for (int kt = 0; kt < SEQN; kt += 32) {
    // Stage K tile (32 keys x 64 d), row-major.
#pragma unroll
    for (int i = 0; i < 2; ++i) {
      const int c = threadIdx.x + 128 * i;      // 256 16B chunks
      const int r = c >> 3, ch = c & 7;
      *(uint4*)&Ks[r * KST + ch * 8] =
          *(const uint4*)(qkv + base + 768 + (size_t)(kt + r) * TC + ch * 8);
    }
    // Stage V tile transposed: Vs[d][key].
#pragma unroll
    for (int i = 0; i < 2; ++i) {
      const int c = threadIdx.x + 128 * i;
      const int r = c >> 3, ch = c & 7;
      const h8 vv = *(const h8*)(qkv + base + 1536 + (size_t)(kt + r) * TC + ch * 8);
#pragma unroll
      for (int j = 0; j < 8; ++j) Vs[(ch * 8 + j) * VST + r] = vv[j];
    }
    __syncthreads();

    // S = Q @ K^T over this key tile (16x32), two 16-col sub-tiles.
    v8f s[2];
#pragma unroll
    for (int ns = 0; ns < 2; ++ns) {
      const int n = ns * 16 + lmod;
      Frag16 kb;
      v8f z;
#pragma unroll
      for (int v = 0; v < 8; ++v) z[v] = 0.0f;
      kb.h[0] = *(const h8*)&Ks[n * KST + klo];
      kb.h[1] = *(const h8*)&Ks[n * KST + klo + 16];
      s[ns] = __builtin_amdgcn_wmma_f32_16x16x32_f16(
          false, q0.v, false, kb.v, (short)0, z, false, false);
      kb.h[0] = *(const h8*)&Ks[n * KST + 32 + klo];
      kb.h[1] = *(const h8*)&Ks[n * KST + 32 + klo + 16];
      s[ns] = __builtin_amdgcn_wmma_f32_16x16x32_f16(
          false, q1.v, false, kb.v, (short)0, s[ns], false, false);
    }

    // Online softmax. Row m = v + 8*lhi lives across the lane's 16-lane half.
    float alpha[8];
#pragma unroll
    for (int v = 0; v < 8; ++v) {
      float s0 = s[0][v], s1 = s[1][v];
      float t = fmaxf(s0, s1);
#pragma unroll
      for (int m = 1; m < 16; m <<= 1) t = fmaxf(t, __shfl_xor(t, m, 32));
      const float mn = fmaxf(mrow[v], t);
      alpha[v] = __expf(mrow[v] - mn);
      mrow[v] = mn;
      const float p0 = __expf(s0 - mn), p1 = __expf(s1 - mn);
      s[0][v] = p0; s[1][v] = p1;
      float rs = p0 + p1;
#pragma unroll
      for (int m = 1; m < 16; m <<= 1) rs += __shfl_xor(rs, m, 32);
      lrow[v] = lrow[v] * alpha[v] + rs;
    }

    // P (C-layout) -> LDS -> A-fragment layout. Wave-private region: DS ops
    // from one wave complete in order, so an in-wave dscnt wait suffices.
#pragma unroll
    for (int v = 0; v < 8; ++v) {
      const int row = v + lhi * 8;
      Ps[wave][row * PST + lmod]      = (f16)s[0][v];
      Ps[wave][row * PST + 16 + lmod] = (f16)s[1][v];
    }
    asm volatile("s_wait_dscnt 0" ::: "memory");

    Frag16 pa;
    pa.h[0] = *(const h8*)&Ps[wave][lmod * PST + klo];
    pa.h[1] = *(const h8*)&Ps[wave][lmod * PST + klo + 16];
#pragma unroll
    for (int os = 0; os < 4; ++os) {
#pragma unroll
      for (int v = 0; v < 8; ++v) o[os][v] *= alpha[v];
      Frag16 vb;
      const int d = os * 16 + lmod;
      vb.h[0] = *(const h8*)&Vs[d * VST + klo];
      vb.h[1] = *(const h8*)&Vs[d * VST + klo + 16];
      o[os] = __builtin_amdgcn_wmma_f32_16x16x32_f16(
          false, pa.v, false, vb.v, (short)0, o[os], false, false);
    }
    __syncthreads();   // before next tile overwrites Ks/Vs
  }

  // Normalize and store: out[(b*N + q)*C + h*64 + d].
#pragma unroll
  for (int os = 0; os < 4; ++os)
#pragma unroll
    for (int v = 0; v < 8; ++v) {
      const int q   = qbase + v + lhi * 8;
      const int col = h * HD + os * 16 + lmod;
      out[((size_t)b * SEQN + q) * DIMC + col] = (f16)(o[os][v] / lrow[v]);
    }
}

// ---------------------------------------------------------------------------
extern "C" void kernel_launch(void* const* d_in, const int* in_sizes, int n_in,
                              void* d_out, int out_size, void* d_ws, size_t ws_size,
                              hipStream_t stream) {
  (void)in_sizes; (void)n_in; (void)out_size; (void)ws_size;
  const float* x      = (const float*)d_in[0];
  const float* w_qkv  = (const float*)d_in[1];
  const float* b_qkv  = (const float*)d_in[2];
  const float* w_proj = (const float*)d_in[3];
  const float* b_proj = (const float*)d_in[4];
  const float* g1     = (const float*)d_in[5];
  const float* beta1  = (const float*)d_in[6];
  const float* g2     = (const float*)d_in[7];
  const float* beta2  = (const float*)d_in[8];
  const float* w_fc1  = (const float*)d_in[9];
  const float* b_fc1  = (const float*)d_in[10];
  const float* w_fc2  = (const float*)d_in[11];
  const float* b_fc2  = (const float*)d_in[12];

  char* ws = (char*)d_ws;
  // Workspace layout (bytes); h overlays dead qkv+attn region.
  f16*   xh   = (f16*)(ws + 0);            // 12,582,912 B  (ln1 out; reused as ln2 out)
  f16*   qkv  = (f16*)(ws + 12582912);     // 37,748,736 B
  f16*   attn = (f16*)(ws + 50331648);     // 12,582,912 B
  float* x2   = (float*)(ws + 62914560);   // 25,165,824 B  (ends 88,080,384)
  f16*   hbuf = (f16*)(ws + 12582912);     // 50,331,648 B  (overlays qkv+attn)
  float* outp = (float*)d_out;

  // 1) LN1
  ln_kernel<<<MROWS, 256, 0, stream>>>(x, g1, beta1, xh);
  // 2) QKV GEMM (8192 x 2304 x 768) -> f16
  gemm_wmma<false, true, false><<<dim3(TC / 128, MROWS / 128), 256, 0, stream>>>(
      xh, w_qkv, b_qkv, nullptr, qkv, nullptr, MROWS, TC, DIMC);
  // 3) Attention
  attn_kernel<<<dim3(SEQN / 64, 8 * NHEAD), 128, 0, stream>>>(qkv, attn);
  // 4) Proj GEMM + residual(x) -> fp32 x2
  gemm_wmma<false, false, true><<<dim3(DIMC / 128, MROWS / 128), 256, 0, stream>>>(
      attn, w_proj, b_proj, x, nullptr, x2, MROWS, DIMC, DIMC);
  // 5) LN2
  ln_kernel<<<MROWS, 256, 0, stream>>>(x2, g2, beta2, xh);
  // 6) FC1 GEMM + GELU -> f16 h
  gemm_wmma<true, true, false><<<dim3(HIDDEN / 128, MROWS / 128), 256, 0, stream>>>(
      xh, w_fc1, b_fc1, nullptr, hbuf, nullptr, MROWS, HIDDEN, DIMC);
  // 7) FC2 GEMM + residual(x2) -> fp32 out
  gemm_wmma<false, false, true><<<dim3(DIMC / 128, MROWS / 128), 256, 0, stream>>>(
      hbuf, w_fc2, b_fc2, x2, nullptr, outp, MROWS, DIMC, HIDDEN);
}
